// Sequencer2DBlock_60232621359575
// MI455X (gfx1250) — matrix operand order repaired and checked by reading, not compiled, not run on hardware
//
#include <hip/hip_runtime.h>
#include <hip/hip_bf16.h>
#include <math.h>

// ---------------------------------------------------------------------------
// Shapes (fixed by the reference): B=4, H=56, W=56, C=192
//   T   = 12544 tokens          di = 384   ds = 16   dr = 12   K(conv) = 4
//   dbc dim = dr + 2*ds = 44    2*di = 768 hid = 576
// ---------------------------------------------------------------------------
#define TOKENS   12544
#define CDIM     192
#define DI       384
#define DS       16
#define DR       12
#define DBC      44
#define SEQN     224      // sequences per axis (B*H or B*W)
#define SEQL     56

typedef __attribute__((ext_vector_type(16))) __bf16 v16bf;
typedef __attribute__((ext_vector_type(8)))  __bf16 v8bf;
typedef __attribute__((ext_vector_type(8)))  float  v8f;

enum { EPI_F32 = 0, EPI_BF16 = 1, EPI_BIAS_RES_F32 = 2, EPI_GELU_BF16 = 3 };

// ---------------------------------------------------------------------------
// fp32 -> bf16 conversion (weights / one-shot)
// ---------------------------------------------------------------------------
__global__ __launch_bounds__(256)
void f2bf_kernel(const float* __restrict__ in, __bf16* __restrict__ out, int n) {
    int i = blockIdx.x * 256 + threadIdx.x;
    if (i < n) out[i] = (__bf16)in[i];
}

// ---------------------------------------------------------------------------
// LayerNorm over C=192, one wave32 per token (6 elems/lane), bf16 output
// ---------------------------------------------------------------------------
__global__ __launch_bounds__(256)
void ln_kernel(const float* __restrict__ x, const float* __restrict__ g,
               const float* __restrict__ b, __bf16* __restrict__ out) {
    int lane = threadIdx.x & 31;
    int tok  = blockIdx.x * 8 + (threadIdx.x >> 5);
    if (tok >= TOKENS) return;
    const float* xp = x + (size_t)tok * CDIM;
    float v[6], s = 0.f, s2 = 0.f;
#pragma unroll
    for (int i = 0; i < 6; ++i) {
        v[i] = xp[lane + i * 32];
        s += v[i]; s2 += v[i] * v[i];
    }
#pragma unroll
    for (int m = 16; m >= 1; m >>= 1) {
        s  += __shfl_xor(s,  m, 32);
        s2 += __shfl_xor(s2, m, 32);
    }
    float mu  = s * (1.0f / CDIM);
    float var = s2 * (1.0f / CDIM) - mu * mu;
    float rs  = rsqrtf(var + 1e-6f);
#pragma unroll
    for (int i = 0; i < 6; ++i) {
        int c = lane + i * 32;
        out[(size_t)tok * CDIM + c] = (__bf16)((v[i] - mu) * rs * g[c] + b[c]);
    }
}

// ---------------------------------------------------------------------------
// bf16 WMMA GEMM:  out[M x N] = A[M x K] * W^T   (W is N x K row-major)
// Block = 128 threads (4 waves). Block tile 128Mx64N; each wave 32Mx64N
// (2 A-fragments x 4 B-tiles = 8 WMMA accumulators; every B fragment is
// shared by two WMMAs -> 12 b128 loads per 8 WMMAs per K-step).
// Fragments loaded directly from global in contiguous k order (k = 16*(lane/16) + s), same for A and B:
//   A: lane<16 -> K chunks {0-7,8-15}; lane>=16 -> {16-23,24-31}  (2x b128)
//   B: lane L  -> col N=L%16, contiguous K16 at +(L/16)*16        (1x b256)
// ---------------------------------------------------------------------------
__global__ __launch_bounds__(128)
void gemm_bf16_kernel(const __bf16* __restrict__ A, const __bf16* __restrict__ Wt,
                      int M, int N, int K, void* __restrict__ outp,
                      int ldc, int col_off, int mode,
                      const float* __restrict__ bias, const float* __restrict__ res) {
    const int lane = threadIdx.x & 31;
    const int wave = threadIdx.x >> 5;
    const int m0   = blockIdx.y * 128 + wave * 32;   // wave covers rows m0..m0+31
    const int nblk = blockIdx.x * 64;
    const int lm   = lane & 15;
    const int half = lane >> 4;

    v8f acc0 = {}, acc1 = {}, acc2 = {}, acc3 = {};
    v8f acc4 = {}, acc5 = {}, acc6 = {}, acc7 = {};

    const __bf16* arow0 = A + (size_t)(m0 + lm) * K + half * 16;        // rows m0..m0+15
    const __bf16* arow1 = A + (size_t)(m0 + 16 + lm) * K + half * 16;   // rows m0+16..m0+31
    for (int k0 = 0; k0 < K; k0 += 32) {
        v8bf a0lo = *(const v8bf*)(arow0 + k0);
        v8bf a0hi = *(const v8bf*)(arow0 + k0 + 8);
        v8bf a1lo = *(const v8bf*)(arow1 + k0);
        v8bf a1hi = *(const v8bf*)(arow1 + k0 + 8);
        v16bf a0 = __builtin_shufflevector(a0lo, a0hi,
                      0, 1, 2, 3, 4, 5, 6, 7, 8, 9, 10, 11, 12, 13, 14, 15);
        v16bf a1 = __builtin_shufflevector(a1lo, a1hi,
                      0, 1, 2, 3, 4, 5, 6, 7, 8, 9, 10, 11, 12, 13, 14, 15);

        const __bf16* bbase = Wt + (size_t)(nblk + lm) * K + k0 + half * 16;
        v16bf b0 = *(const v16bf*)(bbase + (size_t)(0 * 16) * K);
        v16bf b1 = *(const v16bf*)(bbase + (size_t)(1 * 16) * K);
        v16bf b2 = *(const v16bf*)(bbase + (size_t)(2 * 16) * K);
        v16bf b3 = *(const v16bf*)(bbase + (size_t)(3 * 16) * K);

        acc0 = __builtin_amdgcn_wmma_f32_16x16x32_bf16(false, a0, false, b0, (short)0, acc0, false, false);
        acc4 = __builtin_amdgcn_wmma_f32_16x16x32_bf16(false, a1, false, b0, (short)0, acc4, false, false);
        acc1 = __builtin_amdgcn_wmma_f32_16x16x32_bf16(false, a0, false, b1, (short)0, acc1, false, false);
        acc5 = __builtin_amdgcn_wmma_f32_16x16x32_bf16(false, a1, false, b1, (short)0, acc5, false, false);
        acc2 = __builtin_amdgcn_wmma_f32_16x16x32_bf16(false, a0, false, b2, (short)0, acc2, false, false);
        acc6 = __builtin_amdgcn_wmma_f32_16x16x32_bf16(false, a1, false, b2, (short)0, acc6, false, false);
        acc3 = __builtin_amdgcn_wmma_f32_16x16x32_bf16(false, a0, false, b3, (short)0, acc3, false, false);
        acc7 = __builtin_amdgcn_wmma_f32_16x16x32_bf16(false, a1, false, b3, (short)0, acc7, false, false);
    }

    // Epilogue. C/D layout: VGPR r -> row m + (lane/16)*8 + r, col n + lane%16
    v8f* accs[8] = { &acc0, &acc1, &acc2, &acc3, &acc4, &acc5, &acc6, &acc7 };
    const int cn = lm;
#pragma unroll
    for (int g = 0; g < 2; ++g) {          // A-fragment group (rows +0 / +16)
        int mr = m0 + g * 16 + half * 8;
#pragma unroll
        for (int j = 0; j < 4; ++j) {      // N tile
            int n = nblk + j * 16 + cn;
            v8f& a = *accs[g * 4 + j];
#pragma unroll
            for (int r = 0; r < 8; ++r) {
                float v = a[r];
                size_t oidx = (size_t)(mr + r) * ldc + col_off + n;
                if (mode == EPI_F32) {
                    ((float*)outp)[oidx] = v;
                } else if (mode == EPI_BF16) {
                    ((__bf16*)outp)[oidx] = (__bf16)v;
                } else if (mode == EPI_BIAS_RES_F32) {
                    ((float*)outp)[oidx] = v + bias[n] + res[oidx];
                } else { // EPI_GELU_BF16
                    float t = v + bias[n];
                    ((__bf16*)outp)[oidx] = (__bf16)(0.5f * t * (1.0f + erff(t * 0.70710678f)));
                }
            }
        }
    }
}

// ---------------------------------------------------------------------------
// token index for (sequence o, position pos) on a given axis
//   axis 0: sequences along W (token stride 1)   o = b*56+h
//   axis 1: sequences along H (token stride 56)  o = b*56+w
// ---------------------------------------------------------------------------
__device__ __forceinline__ int tok_of(int o, int pos, int axis) {
    return (axis == 0) ? o * SEQL + pos
                       : (o / SEQL) * (SEQL * SEQL) + pos * SEQL + (o % SEQL);
}

// ---------------------------------------------------------------------------
// Causal depthwise conv (K=4) + SiLU. dir=0 forward, dir=1 = conv of flipped
// sequence written back at original positions (taps reversed, rightward).
// Reads the xi half (first DI cols) of xz (T x 768).
// ---------------------------------------------------------------------------
__global__ __launch_bounds__(256)
void conv_silu_kernel(const float* __restrict__ xz, const float* __restrict__ cw,
                      const float* __restrict__ cb, float* __restrict__ xc,
                      int axis, int dir) {
    int idx = blockIdx.x * 256 + threadIdx.x;
    if (idx >= SEQN * SEQL * DI) return;
    int c   = idx % DI;
    int pos = (idx / DI) % SEQL;
    int o   = idx / (DI * SEQL);
    float s = cb[c];
    if (dir == 0) {
#pragma unroll
        for (int k = 0; k < 4; ++k) {
            int p = pos - 3 + k;
            if (p >= 0) s += xz[(size_t)tok_of(o, p, axis) * (2 * DI) + c] * cw[c * 4 + k];
        }
    } else {
#pragma unroll
        for (int j = 0; j < 4; ++j) {
            int p = pos + j;
            if (p < SEQL) s += xz[(size_t)tok_of(o, p, axis) * (2 * DI) + c] * cw[c * 4 + 3 - j];
        }
    }
    xc[(size_t)tok_of(o, pos, axis) * DI + c] = s / (1.0f + __expf(-s));
}

// ---------------------------------------------------------------------------
// Per-token: dbc = Wx @ xc  (44 x 384), dt = softplus(Wdt @ dbc[:12] + bdt)
// One block per token.
// ---------------------------------------------------------------------------
__global__ __launch_bounds__(128)
void dbc_dt_kernel(const float* __restrict__ xc, const float* __restrict__ Wx,
                   const float* __restrict__ Wdt, const float* __restrict__ bdt,
                   float* __restrict__ dbc, float* __restrict__ dt) {
    __shared__ float sxc[DI];
    __shared__ float sdbc[DBC];
    int tok = blockIdx.x;
    int tid = threadIdx.x;
    for (int i = tid; i < DI; i += 128) sxc[i] = xc[(size_t)tok * DI + i];
    __syncthreads();
    if (tid < DBC) {
        const float* wr = Wx + (size_t)tid * DI;
        float a = 0.f;
        for (int j = 0; j < DI; ++j) a += wr[j] * sxc[j];
        sdbc[tid] = a;
        dbc[(size_t)tok * DBC + tid] = a;
    }
    __syncthreads();
    for (int c = tid; c < DI; c += 128) {
        float s = bdt[c];
#pragma unroll
        for (int r = 0; r < DR; ++r) s += sdbc[r] * Wdt[c * DR + r];
        dt[(size_t)tok * DI + c] = (s > 20.f) ? s : log1pf(__expf(s));
    }
}

// ---------------------------------------------------------------------------
// Selective scan + D-skip + z-gating, fused; writes bf16 y for the Wout GEMM.
// One block per sequence (384 threads = one channel each); B/C staged in LDS.
// dir=0: left->right, dir=1: right->left (flipped scan, original positions).
// ---------------------------------------------------------------------------
__global__ __launch_bounds__(384)
void scan_kernel(const float* __restrict__ xc, const float* __restrict__ dt,
                 const float* __restrict__ dbc, const float* __restrict__ xz,
                 const float* __restrict__ Alog, const float* __restrict__ Dp,
                 __bf16* __restrict__ ybf, int axis, int dir) {
    __shared__ float sB[DS];
    __shared__ float sC[DS];
    int o = blockIdx.x;
    int c = threadIdx.x;
    float A[DS], h[DS];
#pragma unroll
    for (int s = 0; s < DS; ++s) {
        A[s] = -__expf(Alog[c * DS + s]);
        h[s] = 0.f;
    }
    float Dc = Dp[c];
    for (int step = 0; step < SEQL; ++step) {
        int pos = (dir == 0) ? step : (SEQL - 1 - step);
        int tok = tok_of(o, pos, axis);
        __syncthreads();
        if (c < 2 * DS) {
            float v = dbc[(size_t)tok * DBC + DR + c];
            if (c < DS) sB[c] = v; else sC[c - DS] = v;
        }
        __syncthreads();
        float d  = dt[(size_t)tok * DI + c];
        float xv = xc[(size_t)tok * DI + c];
        float dx = d * xv;
        float y = 0.f;
#pragma unroll
        for (int s = 0; s < DS; ++s) {
            h[s] = __expf(d * A[s]) * h[s] + dx * sB[s];
            y += h[s] * sC[s];
        }
        y += Dc * xv;
        float z = xz[(size_t)tok * (2 * DI) + DI + c];
        y *= z / (1.0f + __expf(-z));          // * silu(z)
        ybf[(size_t)tok * DI + c] = (__bf16)y;
    }
}

// ---------------------------------------------------------------------------
// Host-side orchestration
// ---------------------------------------------------------------------------
extern "C" void kernel_launch(void* const* d_in, const int* in_sizes, int n_in,
                              void* d_out, int out_size, void* d_ws, size_t ws_size,
                              hipStream_t stream) {
    (void)in_sizes; (void)n_in; (void)out_size; (void)ws_size;
    const int T = TOKENS;

    char* ws = (char*)d_ws;
    size_t off = 0;
    auto alloc = [&](size_t bytes) -> void* {
        void* p = ws + off;
        off = (off + bytes + 255) & ~(size_t)255;
        return p;
    };

    __bf16* xn_bf  = (__bf16*)alloc((size_t)T * CDIM * 2);
    float*  xz     = (float*) alloc((size_t)T * 2 * DI * 4);
    float*  xc     = (float*) alloc((size_t)T * DI * 4);
    float*  dbc    = (float*) alloc((size_t)T * DBC * 4);
    float*  dt     = (float*) alloc((size_t)T * DI * 4);
    __bf16* y_bf   = (__bf16*)alloc((size_t)T * DI * 2);
    __bf16* u_bf   = (__bf16*)alloc((size_t)T * 4 * CDIM * 2);
    float*  x1     = (float*) alloc((size_t)T * CDIM * 4);
    __bf16* xn2_bf = (__bf16*)alloc((size_t)T * CDIM * 2);
    __bf16* m_bf   = (__bf16*)alloc((size_t)T * 576 * 2);
    __bf16* Win_bf[2]  = { (__bf16*)alloc(768 * 192 * 2), (__bf16*)alloc(768 * 192 * 2) };
    __bf16* Wout_bf[2] = { (__bf16*)alloc(192 * 384 * 2), (__bf16*)alloc(192 * 384 * 2) };
    __bf16* fcw_bf = (__bf16*)alloc(192 * 768 * 2);
    __bf16* w1_bf  = (__bf16*)alloc(576 * 192 * 2);
    __bf16* w2_bf  = (__bf16*)alloc(192 * 576 * 2);

    auto cvt = [&](const void* src, __bf16* dst, int n) {
        hipLaunchKernelGGL(f2bf_kernel, dim3((n + 255) / 256), dim3(256), 0, stream,
                           (const float*)src, dst, n);
    };
    auto gemm = [&](const __bf16* A, const __bf16* Wt, int N, int K, void* out,
                    int ldc, int coff, int mode, const float* bias, const float* res) {
        hipLaunchKernelGGL(gemm_bf16_kernel, dim3(N / 64, T / 128), dim3(128), 0, stream,
                           A, Wt, T, N, K, out, ldc, coff, mode, bias, res);
    };

    // weight conversions (axis 0 = mh @ d_in[5..13], axis 1 = mv @ d_in[14..22])
    cvt(d_in[5],  Win_bf[0],  768 * 192);
    cvt(d_in[13], Wout_bf[0], 192 * 384);
    cvt(d_in[14], Win_bf[1],  768 * 192);
    cvt(d_in[22], Wout_bf[1], 192 * 384);
    cvt(d_in[23], fcw_bf,     192 * 768);
    cvt(d_in[25], w1_bf,      576 * 192);
    cvt(d_in[27], w2_bf,      192 * 576);

    // LN1
    hipLaunchKernelGGL(ln_kernel, dim3(T / 8), dim3(256), 0, stream,
                       (const float*)d_in[0], (const float*)d_in[1],
                       (const float*)d_in[2], xn_bf);

    const int conv_blocks = (SEQN * SEQL * DI + 255) / 256;
    for (int axis = 0; axis < 2; ++axis) {
        int base = 5 + axis * 9;
        const float* convw = (const float*)d_in[base + 1];
        const float* convb = (const float*)d_in[base + 2];
        const float* Wx    = (const float*)d_in[base + 3];
        const float* Wdt   = (const float*)d_in[base + 4];
        const float* bdt   = (const float*)d_in[base + 5];
        const float* Alog  = (const float*)d_in[base + 6];
        const float* Dp    = (const float*)d_in[base + 7];

        // xz = xn @ Win^T   (T x 768)
        gemm(xn_bf, Win_bf[axis], 768, 192, xz, 768, 0, EPI_F32, nullptr, nullptr);

        for (int dir = 0; dir < 2; ++dir) {
            hipLaunchKernelGGL(conv_silu_kernel, dim3(conv_blocks), dim3(256), 0, stream,
                               xz, convw, convb, xc, axis, dir);
            hipLaunchKernelGGL(dbc_dt_kernel, dim3(T), dim3(128), 0, stream,
                               xc, Wx, Wdt, bdt, dbc, dt);
            hipLaunchKernelGGL(scan_kernel, dim3(SEQN), dim3(DI), 0, stream,
                               xc, dt, dbc, xz, Alog, Dp, y_bf, axis, dir);
            // u = concat([v_fwd, v_bwd, h_fwd, h_bwd]); mh(axis0)->cols 384+, mv->0+
            int coff = (axis == 0 ? 2 * CDIM : 0) + dir * CDIM;
            gemm(y_bf, Wout_bf[axis], 192, 384, u_bf, 4 * CDIM, coff,
                 EPI_BF16, nullptr, nullptr);
        }
    }

    // x1 = x + u @ fc_w^T + fc_b
    gemm(u_bf, fcw_bf, 192, 768, x1, 192, 0, EPI_BIAS_RES_F32,
         (const float*)d_in[24], (const float*)d_in[0]);
    // LN2
    hipLaunchKernelGGL(ln_kernel, dim3(T / 8), dim3(256), 0, stream,
                       x1, (const float*)d_in[3], (const float*)d_in[4], xn2_bf);
    // m = gelu(xn2 @ w1^T + b1)
    gemm(xn2_bf, w1_bf, 576, 192, m_bf, 576, 0, EPI_GELU_BF16,
         (const float*)d_in[26], nullptr);
    // out = x1 + m @ w2^T + b2
    gemm(m_bf, w2_bf, 192, 576, d_out, 192, 0, EPI_BIAS_RES_F32,
         (const float*)d_in[28], x1);
}
